// GCN_Unit_50852412784969
// MI455X (gfx1250) — matrix operand order repaired and checked by reading, not compiled, not run on hardware
//
#include <hip/hip_runtime.h>
#include <hip/hip_bf16.h>
#include <math.h>

typedef __attribute__((ext_vector_type(16))) _Float16 v16h;
typedef __attribute__((ext_vector_type(8)))  float    v8f;

#define ALPHA 0.2f
#define NNODE 4096
#define NBATCH 2
#define KNN 7

union FragH { v16h v; unsigned u[8]; };

static __device__ __forceinline__ bool better(float v1, int i1, float v2, int i2) {
    return (v1 < v2) || (v1 == v2 && i1 < i2);
}

// ---------------------------------------------------------------------------
// f32 -> f16 convert (weights)
// ---------------------------------------------------------------------------
__global__ void f32_to_f16_kernel(const float* __restrict__ src,
                                  unsigned short* __restrict__ dst, int n) {
    int i = blockIdx.x * blockDim.x + threadIdx.x;
    if (i < n) {
        _Float16 h = (_Float16)src[i];
        dst[i] = *(unsigned short*)&h;
    }
}

// ---------------------------------------------------------------------------
// Head conv: 31->64, k=3, s=2, p=1.  Writes node-major f16 features [n][c]
// and per-node squared norm.  One block per node, 64 threads (one per co).
// ---------------------------------------------------------------------------
__global__ void conv_head_kernel(const float* __restrict__ x,
                                 const float* __restrict__ w_head,
                                 const float* __restrict__ b_head,
                                 unsigned short* __restrict__ ynh,
                                 float* __restrict__ sq) {
    __shared__ float sp[31 * 9];
    __shared__ float ssq[64];
    int n  = blockIdx.x;           // 0..8191 (global node)
    int b  = n >> 12;
    int nn = n & 4095;
    int oy = nn >> 6, ox = nn & 63;
    for (int l = threadIdx.x; l < 31 * 9; l += 64) {
        int ci = l / 9, kk = l % 9, ky = kk / 3, kx = kk % 3;
        int iy = 2 * oy + ky - 1, ix = 2 * ox + kx - 1;
        float v = 0.f;
        if (iy >= 0 && iy < 128 && ix >= 0 && ix < 128)
            v = x[((size_t)(b * 31 + ci) * 128 + iy) * 128 + ix];
        sp[l] = v;
    }
    __syncthreads();
    int co = threadIdx.x;
    float acc = b_head[co];
    for (int ci = 0; ci < 31; ++ci) {
        #pragma unroll
        for (int kk = 0; kk < 9; ++kk)
            acc = fmaf(sp[ci * 9 + kk], w_head[(size_t)(co * 31 + ci) * 9 + kk], acc);
    }
    _Float16 hv = (_Float16)acc;
    ynh[(size_t)n * 64 + co] = *(unsigned short*)&hv;
    ssq[co] = acc * acc;
    __syncthreads();
    if (co == 0) {
        float s = 0.f;
        for (int i = 0; i < 64; ++i) s += ssq[i];
        sq[n] = s;
    }
}

// ---------------------------------------------------------------------------
// Fused kNN: WMMA Gram tiles + running top-7 per row.  One block per 16 rows,
// 8 waves; wave w covers column tiles jt = w + 8t (512 cols each).
// d2(i,j) = |yi|^2 + |yj|^2 - 2 <yi,yj>;  sqrt is monotone -> skipped.
// Tie-break (value, index) lexicographic == stable argsort ranks.
// ---------------------------------------------------------------------------
__global__ void knn_kernel(const unsigned short* __restrict__ ynh,
                           const float* __restrict__ sq,
                           int* __restrict__ nbr) {
    __shared__ float sD[8][256];
    __shared__ float sVal[8][16][KNN];
    __shared__ int   sIdx[8][16][KNN];
    const int tid  = threadIdx.x;
    const int w    = tid >> 5;
    const int lane = tid & 31;
    const int hh   = lane >> 4;
    const int r    = lane & 15;
    const int R0   = blockIdx.x * 16;      // global row base
    const int b    = R0 >> 12;
    const int colBase = b << 12;           // first node of this batch

    // A fragments (rows R0..R0+15, K = 0..63), documented 16-bit A layout.
    FragH A0, A1;
    const unsigned* arow = (const unsigned*)(ynh + (size_t)(R0 + r) * 64);
    #pragma unroll
    for (int v = 0; v < 8; ++v) {
        int kOff = ((v & 4) << 2) + (hh << 3) + ((v & 3) << 1);
        A0.u[v] = arow[kOff >> 1];
        A1.u[v] = arow[(32 + kOff) >> 1];
    }
    // row |y|^2 per accumulator slot (fixed across tiles)
    float rsq[8];
    #pragma unroll
    for (int v = 0; v < 8; ++v) rsq[v] = sq[R0 + (hh << 3) + v];

    float tv[KNN]; int ti[KNN];
    #pragma unroll
    for (int q = 0; q < KNN; ++q) { tv[q] = 3.0e38f; ti[q] = 0x7fffffff; }

    for (int t = 0; t < 32; ++t) {
        int jt = w + (t << 3);
        int j0 = jt << 4;                  // within-batch column base
        const unsigned* bcol =
            (const unsigned*)(ynh + (size_t)(colBase + j0 + r) * 64);
        FragH B0, B1;
        #pragma unroll
        for (int v = 0; v < 8; ++v) {
            int k = ((v & 4) << 2) + (hh << 3) + ((v & 3) << 1);  // two packed K's are consecutive channels
            B0.u[v] = bcol[k >> 1];
            B1.u[v] = bcol[(32 + k) >> 1];
        }
        v8f acc = {};
        acc = __builtin_amdgcn_wmma_f32_16x16x32_f16(false, A0.v, false, B0.v,
                                                     (short)0, acc, false, false);
        acc = __builtin_amdgcn_wmma_f32_16x16x32_f16(false, A1.v, false, B1.v,
                                                     (short)0, acc, false, false);
        float sqc = sq[colBase + j0 + r];
        #pragma unroll
        for (int v = 0; v < 8; ++v) {
            int m = (hh << 3) + v;
            sD[w][m * 16 + r] = rsq[v] + sqc - 2.0f * acc[v];
        }
        __syncthreads();
        if (hh == 0) {                     // lanes 0-15 own one row each
            for (int cc = 0; cc < 16; ++cc) {
                float dv = sD[w][r * 16 + cc];
                int   di = j0 + cc;
                if (better(dv, di, tv[KNN - 1], ti[KNN - 1])) {
                    tv[KNN - 1] = dv; ti[KNN - 1] = di;
                    #pragma unroll
                    for (int q = KNN - 2; q >= 0; --q) {
                        if (better(tv[q + 1], ti[q + 1], tv[q], ti[q])) {
                            float fv = tv[q]; int fi = ti[q];
                            tv[q] = tv[q + 1]; ti[q] = ti[q + 1];
                            tv[q + 1] = fv;   ti[q + 1] = fi;
                        }
                    }
                }
            }
        }
        __syncthreads();
    }
    if (hh == 0) {
        #pragma unroll
        for (int q = 0; q < KNN; ++q) { sVal[w][r][q] = tv[q]; sIdx[w][r][q] = ti[q]; }
    }
    __syncthreads();
    if (tid < 16) {                        // merge 8 partial top-7 lists per row
        float fv[KNN]; int fi[KNN];
        #pragma unroll
        for (int q = 0; q < KNN; ++q) { fv[q] = 3.0e38f; fi[q] = 0x7fffffff; }
        for (int ww = 0; ww < 8; ++ww)
            for (int q = 0; q < KNN; ++q) {
                float dv = sVal[ww][tid][q]; int di = sIdx[ww][tid][q];
                if (better(dv, di, fv[KNN - 1], fi[KNN - 1])) {
                    fv[KNN - 1] = dv; fi[KNN - 1] = di;
                    #pragma unroll
                    for (int q2 = KNN - 2; q2 >= 0; --q2) {
                        if (better(fv[q2 + 1], fi[q2 + 1], fv[q2], fi[q2])) {
                            float a = fv[q2]; int c = fi[q2];
                            fv[q2] = fv[q2 + 1]; fi[q2] = fi[q2 + 1];
                            fv[q2 + 1] = a;      fi[q2 + 1] = c;
                        }
                    }
                }
            }
        #pragma unroll
        for (int q = 0; q < KNN; ++q) nbr[(size_t)(R0 + tid) * 8 + q] = fi[q];
        nbr[(size_t)(R0 + tid) * 8 + 7] = 0;
    }
}

// ---------------------------------------------------------------------------
// WMMA GEMM: Out[M x 64] (f32) = Xh[M x K] (f16, K contiguous) * Wh[K x 64].
// Block = 4 waves; wave w does cols [16w,16w+16); block does 16 rows.
// ---------------------------------------------------------------------------
__global__ void gemm_wmma_kernel(const unsigned short* __restrict__ Xh,
                                 const unsigned short* __restrict__ Wh,
                                 float* __restrict__ Out, int K) {
    const int tid  = threadIdx.x;
    const int w    = tid >> 5;
    const int lane = tid & 31;
    const int hh   = lane >> 4;
    const int r    = lane & 15;
    const int m0   = blockIdx.x * 16;
    const int n0   = w * 16;
    v8f acc = {};
    for (int kb = 0; kb < K; kb += 32) {
        FragH A, Bf;
        const unsigned* arow = (const unsigned*)(Xh + (size_t)(m0 + r) * K + kb);
        #pragma unroll
        for (int v = 0; v < 8; ++v) {
            int kOff = ((v & 4) << 2) + (hh << 3) + ((v & 3) << 1);
            A.u[v] = arow[kOff >> 1];
            int k = kb + ((v & 4) << 2) + (hh << 3) + ((v & 3) << 1);
            unsigned lo = Wh[(size_t)k * 64 + n0 + r];
            unsigned hi = Wh[(size_t)(k + 1) * 64 + n0 + r];
            Bf.u[v] = lo | (hi << 16);
        }
        acc = __builtin_amdgcn_wmma_f32_16x16x32_f16(false, A.v, false, Bf.v,
                                                     (short)0, acc, false, false);
    }
    #pragma unroll
    for (int v = 0; v < 8; ++v)
        Out[(size_t)(m0 + (hh << 3) + v) * 64 + n0 + r] = acc[v];
}

// ---------------------------------------------------------------------------
// Attention scores: f1[i] = a[:64].h_i ; f2[i] = a[64:].h_i
// ---------------------------------------------------------------------------
__global__ void scores_kernel(const float* __restrict__ H,
                              const float* __restrict__ a,
                              float* __restrict__ f1, float* __restrict__ f2) {
    int i = blockIdx.x * blockDim.x + threadIdx.x;
    if (i >= NBATCH * NNODE) return;
    const float* h = H + (size_t)i * 64;
    float s1 = 0.f, s2 = 0.f;
    #pragma unroll 4
    for (int c = 0; c < 64; ++c) {
        s1 = fmaf(h[c], a[c], s1);
        s2 = fmaf(h[c], a[64 + c], s2);
    }
    f1[i] = s1; f2[i] = s2;
}

// ---------------------------------------------------------------------------
// Sparse GAT aggregation over the 7 neighbors: softmax(leakyrelu(f1_i+f2_j))
// then hp_i = sum att_j * h_j, optional ReLU.  256 thr = 4 nodes x 64 ch.
// ---------------------------------------------------------------------------
__global__ void aggregate_kernel(const float* __restrict__ H,
                                 const float* __restrict__ f1,
                                 const float* __restrict__ f2,
                                 const int* __restrict__ nbr,
                                 float* __restrict__ outF,
                                 unsigned short* __restrict__ outH,
                                 int outStride, int outOff, int doRelu) {
    __shared__ float se[4][KNN];
    __shared__ int   sj[4][KNN];
    int g = threadIdx.x >> 6;
    int c = threadIdx.x & 63;
    int i = blockIdx.x * 4 + g;
    int base = (i >> 12) << 12;
    if (c < KNN) {
        int jg = base + nbr[(size_t)i * 8 + c];
        float e = f1[i] + f2[jg];
        se[g][c] = e > 0.f ? e : ALPHA * e;
        sj[g][c] = jg;
    }
    __syncthreads();
    float m = se[g][0];
    #pragma unroll
    for (int t = 1; t < KNN; ++t) m = fmaxf(m, se[g][t]);
    float s = 0.f, acc = 0.f;
    #pragma unroll
    for (int t = 0; t < KNN; ++t) {
        float wgt = __expf(se[g][t] - m);
        s += wgt;
        acc = fmaf(wgt, H[(size_t)sj[g][t] * 64 + c], acc);
    }
    acc /= s;
    if (doRelu) acc = fmaxf(acc, 0.f);
    if (outF) outF[(size_t)i * 64 + c] = acc;
    if (outH) {
        _Float16 hv = (_Float16)acc;
        outH[(size_t)i * outStride + outOff + c] = *(unsigned short*)&hv;
    }
}

// ---------------------------------------------------------------------------
// elu + log_softmax over 64 channels; writes channel-major z[b][c][n]
// ---------------------------------------------------------------------------
__global__ void elu_lsm_kernel(const float* __restrict__ hpo,
                               float* __restrict__ z) {
    __shared__ float sv[4][64];
    int g = threadIdx.x >> 6, c = threadIdx.x & 63;
    int i = blockIdx.x * 4 + g;
    int b = i >> 12, n = i & 4095;
    float v = hpo[(size_t)i * 64 + c];
    v = v > 0.f ? v : (__expf(v) - 1.f);
    sv[g][c] = v;
    __syncthreads();
    float m = sv[g][0];
    for (int t = 1; t < 64; ++t) m = fmaxf(m, sv[g][t]);
    float s = 0.f;
    for (int t = 0; t < 64; ++t) s += __expf(sv[g][t] - m);
    z[(size_t)(b * 64 + c) * 4096 + n] = v - m - __logf(s);
}

// ---------------------------------------------------------------------------
// ConvTranspose2d(64->64, k=3, s=2, p=1, op=1), gather form:
// out[co,oy,ox] = b[co] + sum z[ci,(oy+1-ky)/2,(ox+1-kx)/2] * w[ci,co,ky,kx]
// ---------------------------------------------------------------------------
__global__ void convt_kernel(const float* __restrict__ z,
                             const float* __restrict__ w_last,
                             const float* __restrict__ b_last,
                             float* __restrict__ out) {
    int t = blockIdx.x * blockDim.x + threadIdx.x;
    if (t >= NBATCH * 64 * 128 * 128) return;
    int ox = t & 127, oy = (t >> 7) & 127, co = (t >> 14) & 63, b = t >> 20;
    float acc = b_last[co];
    #pragma unroll
    for (int ky = 0; ky < 3; ++ky) {
        int ty = oy + 1 - ky;
        if (ty < 0 || (ty & 1)) continue;
        int iy = ty >> 1;
        if (iy >= 64) continue;
        #pragma unroll
        for (int kx = 0; kx < 3; ++kx) {
            int tx = ox + 1 - kx;
            if (tx < 0 || (tx & 1)) continue;
            int ix = tx >> 1;
            if (ix >= 64) continue;
            const float* zp = z + (size_t)b * 64 * 4096 + iy * 64 + ix;
            const float* wp = w_last + co * 9 + ky * 3 + kx;
            #pragma unroll 4
            for (int ci = 0; ci < 64; ++ci)
                acc = fmaf(zp[(size_t)ci * 4096], wp[(size_t)ci * 576], acc);
        }
    }
    out[t] = acc;
}

// ---------------------------------------------------------------------------
extern "C" void kernel_launch(void* const* d_in, const int* in_sizes, int n_in,
                              void* d_out, int out_size, void* d_ws, size_t ws_size,
                              hipStream_t stream) {
    (void)in_sizes; (void)n_in; (void)out_size; (void)ws_size;
    const float* x      = (const float*)d_in[0];
    const float* w_head = (const float*)d_in[1];
    const float* b_head = (const float*)d_in[2];
    const float* W0     = (const float*)d_in[3];
    const float* a0     = (const float*)d_in[4];
    const float* W1     = (const float*)d_in[5];
    const float* a1     = (const float*)d_in[6];
    const float* W_out  = (const float*)d_in[7];
    const float* a_out  = (const float*)d_in[8];
    const float* w_last = (const float*)d_in[9];
    const float* b_last = (const float*)d_in[10];

    const int M = NBATCH * NNODE;   // 8192 nodes
    char* ws = (char*)d_ws;
    size_t off = 0;
    auto alloc = [&](size_t bytes) -> char* {
        char* p = ws + off;
        off = (off + bytes + 255) & ~(size_t)255;
        return p;
    };
    unsigned short* ynh  = (unsigned short*)alloc((size_t)M * 64 * 2);
    float*          sq   = (float*)         alloc((size_t)M * 4);
    int*            nbr  = (int*)           alloc((size_t)M * 8 * 4);
    unsigned short* W0h  = (unsigned short*)alloc(64 * 64 * 2);
    unsigned short* W1h  = (unsigned short*)alloc(64 * 64 * 2);
    unsigned short* Woh  = (unsigned short*)alloc(128 * 64 * 2);
    float*          h0   = (float*)         alloc((size_t)M * 64 * 4);
    float*          h1   = (float*)         alloc((size_t)M * 64 * 4);
    float*          f1_0 = (float*)         alloc((size_t)M * 4);
    float*          f2_0 = (float*)         alloc((size_t)M * 4);
    float*          f1_1 = (float*)         alloc((size_t)M * 4);
    float*          f2_1 = (float*)         alloc((size_t)M * 4);
    float*          f1_o = (float*)         alloc((size_t)M * 4);
    float*          f2_o = (float*)         alloc((size_t)M * 4);
    unsigned short* hcat = (unsigned short*)alloc((size_t)M * 128 * 2);
    float*          ho   = (float*)         alloc((size_t)M * 64 * 4);
    float*          hpo  = (float*)         alloc((size_t)M * 64 * 4);
    float*          zbuf = (float*)         alloc((size_t)M * 64 * 4);

    // weight converts
    f32_to_f16_kernel<<<(64 * 64 + 255) / 256, 256, 0, stream>>>(W0, W0h, 64 * 64);
    f32_to_f16_kernel<<<(64 * 64 + 255) / 256, 256, 0, stream>>>(W1, W1h, 64 * 64);
    f32_to_f16_kernel<<<(128 * 64 + 255) / 256, 256, 0, stream>>>(W_out, Woh, 128 * 64);

    // head conv -> node features (f16) + squared norms
    conv_head_kernel<<<M, 64, 0, stream>>>(x, w_head, b_head, ynh, sq);

    // fused WMMA Gram + top-7 neighbor selection
    knn_kernel<<<M / 16, 256, 0, stream>>>(ynh, sq, nbr);

    // GAT heads
    gemm_wmma_kernel<<<M / 16, 128, 0, stream>>>(ynh, W0h, h0, 64);
    gemm_wmma_kernel<<<M / 16, 128, 0, stream>>>(ynh, W1h, h1, 64);
    scores_kernel<<<(M + 255) / 256, 256, 0, stream>>>(h0, a0, f1_0, f2_0);
    scores_kernel<<<(M + 255) / 256, 256, 0, stream>>>(h1, a1, f1_1, f2_1);
    aggregate_kernel<<<M / 4, 256, 0, stream>>>(h0, f1_0, f2_0, nbr,
                                                nullptr, hcat, 128, 0, 1);
    aggregate_kernel<<<M / 4, 256, 0, stream>>>(h1, f1_1, f2_1, nbr,
                                                nullptr, hcat, 128, 64, 1);

    // output GAT layer (K = 128)
    gemm_wmma_kernel<<<M / 16, 128, 0, stream>>>(hcat, Woh, ho, 128);
    scores_kernel<<<(M + 255) / 256, 256, 0, stream>>>(ho, a_out, f1_o, f2_o);
    aggregate_kernel<<<M / 4, 256, 0, stream>>>(ho, f1_o, f2_o, nbr,
                                                hpo, nullptr, 64, 0, 0);

    // elu + log_softmax -> channel-major z
    elu_lsm_kernel<<<M / 4, 256, 0, stream>>>(hpo, zbuf);

    // transposed conv -> output
    convt_kernel<<<(NBATCH * 64 * 128 * 128 + 255) / 256, 256, 0, stream>>>(
        zbuf, w_last, b_last, (float*)d_out);
}
